// PFA_27479200760086
// MI455X (gfx1250) — matrix operand order repaired and checked by reading, not compiled, not run on hardware
//
#include <hip/hip_runtime.h>
#include <hip/hip_bf16.h>

#define B_ 2
#define N_ 5000
#define C_ 768
#define H_ 12
#define D_ 64
#define HD_ 768
#define E_ 160000
#define ETOT_ (E_ + N_)       // 165000
#define M_ (B_ * N_)          // 10000 rows for the GEMM
#define KSTEPS_ (C_ / 32)     // 24
#define NTILES_ (HD_ / 16)    // 48
#define NEG_SLOPE_ 0.2f

typedef __bf16 bf16_t;
typedef __attribute__((ext_vector_type(16))) __bf16 v16bf;
typedef __attribute__((ext_vector_type(8)))  __bf16 v8bf;
typedef __attribute__((ext_vector_type(8)))  float  v8f;

// ---- monotonic uint encoding of float for integer atomicMax ----
__device__ __forceinline__ unsigned f2ord(float f) {
    unsigned u = __float_as_uint(f);
    return (u & 0x80000000u) ? ~u : (u | 0x80000000u);
}
__device__ __forceinline__ float ord2f(unsigned u) {
    return (u & 0x80000000u) ? __uint_as_float(u & 0x7fffffffu)
                             : __uint_as_float(~u);
}
// f2ord(-inf) = ~0xFF800000 = 0x007FFFFF
#define ORD_NEG_INF 0x007FFFFFu

__device__ __forceinline__ float lrelu(float v) {
    return v > 0.0f ? v : NEG_SLOPE_ * v;
}

// ---------------- kernel 1a: f32 -> bf16 convert (x) ----------------
__global__ void k_cvt_bf16(const float* __restrict__ in, bf16_t* __restrict__ out, int n) {
    int i = blockIdx.x * blockDim.x + threadIdx.x;
    if (i < n) out[i] = (bf16_t)in[i];
}

// ---------------- kernel 1b: convert + pack W into fragment layout ----------------
// Wp[((tn*KSTEPS_ + ks)*32 + lane)*16 + i] = bf16(W[(ks*32 + (i<8 ? hi*8 + i : 16 + hi*8 + i-8))*HD_ + tn*16 + lr])
// so each lane's B-fragment for (tn, ks) is 32 contiguous bytes.
__global__ void k_pack_w(const float* __restrict__ W, bf16_t* __restrict__ Wp) {
    int idx = blockIdx.x * blockDim.x + threadIdx.x;   // over C_*HD_ = 589824
    if (idx >= C_ * HD_) return;
    int i    = idx & 15;
    int lane = (idx >> 4) & 31;
    int ks   = (idx >> 9) % KSTEPS_;
    int tn   = idx / (KSTEPS_ * 512);
    int lr = lane & 15, hi = lane >> 4;
    int k   = ks * 32 + (i < 8 ? hi * 8 + i : 16 + hi * 8 + (i - 8));
    int col = tn * 16 + lr;
    Wp[idx] = (bf16_t)W[k * HD_ + col];
}

// ---------------- kernel 2: WMMA GEMM  h = x @ W ----------------
// A: [M_ x C_] bf16 row-major; Wp: packed fragments; Hout: [M_ x HD_] f32.
// Each wave computes a 32x64 block (2 M-tiles x 4 N-tiles): per k-step,
// 4 A-loads + 8 B-loads (b128) feed 8 WMMAs (1.5 VMEM per WMMA).
__global__ __launch_bounds__(128) void k_gemm_wmma(
    const bf16_t* __restrict__ A, const bf16_t* __restrict__ Wp,
    float* __restrict__ Hout)
{
    const int bm   = blockIdx.x;                      // 0..312 (32-row blocks)
    const int tg   = blockIdx.y * 4 + threadIdx.y;    // 0..11  (groups of 4 N-tiles)
    const int lane = threadIdx.x;                     // 0..31
    const int lr   = lane & 15;
    const int hi   = lane >> 4;

    const int rowA0 = bm * 32 + lr;                   // M-tile 0
    const int rowA1 = rowA0 + 16;                     // M-tile 1
    const int tn0   = tg * 4;

    v8f acc[2][4];
#pragma unroll
    for (int mi = 0; mi < 2; ++mi)
#pragma unroll
        for (int ni = 0; ni < 4; ++ni) acc[mi][ni] = (v8f){};

    // v16bf-granular pointer into packed W; stride between N-tiles = KSTEPS_*32 frags
    const v16bf* __restrict__ WpF =
        (const v16bf*)Wp + (size_t)tn0 * KSTEPS_ * 32 + lane;

    for (int ks = 0; ks < KSTEPS_; ++ks) {
        const int k0 = ks * 32;

        // A fragments (16x32 bf16 each): two contiguous 16B loads per lane
        v16bf af[2];
        {
            const v8bf a00 = *(const v8bf*)(A + rowA0 * C_ + k0 + hi * 8);
            const v8bf a01 = *(const v8bf*)(A + rowA0 * C_ + k0 + 16 + hi * 8);
            const v8bf a10 = *(const v8bf*)(A + rowA1 * C_ + k0 + hi * 8);
            const v8bf a11 = *(const v8bf*)(A + rowA1 * C_ + k0 + 16 + hi * 8);
#pragma unroll
            for (int i = 0; i < 8; ++i) {
                af[0][i] = a00[i]; af[0][8 + i] = a01[i];
                af[1][i] = a10[i]; af[1][8 + i] = a11[i];
            }
        }

        // B fragments: one 32B contiguous load per N-tile
        const v16bf* pb = WpF + (size_t)ks * 32;
        v16bf bfr[4];
#pragma unroll
        for (int ni = 0; ni < 4; ++ni) bfr[ni] = pb[(size_t)ni * KSTEPS_ * 32];

#pragma unroll
        for (int ni = 0; ni < 4; ++ni) {
            acc[0][ni] = __builtin_amdgcn_wmma_f32_16x16x32_bf16(
                             false, af[0], false, bfr[ni], (short)0, acc[0][ni], false, false);
            acc[1][ni] = __builtin_amdgcn_wmma_f32_16x16x32_bf16(
                             false, af[1], false, bfr[ni], (short)0, acc[1][ni], false, false);
        }
    }

    // C/D layout: VGPR j -> M = mtile*16 + hi*8 + j, N = tn*16 + lr
#pragma unroll
    for (int mi = 0; mi < 2; ++mi) {
        const int rbase = bm * 32 + mi * 16 + hi * 8;
#pragma unroll
        for (int j = 0; j < 8; ++j) {
            const int row = rbase + j;
            if (row < M_) {
                float* outb = Hout + row * HD_ + tn0 * 16 + lr;
                outb[ 0] = acc[mi][0][j];
                outb[16] = acc[mi][1][j];
                outb[32] = acc[mi][2][j];
                outb[48] = acc[mi][3][j];
            }
        }
    }
}

// ---------------- kernel 3: attention scores ----------------
__global__ void k_scores(const float* __restrict__ Hm,
                         const float* __restrict__ att_src,
                         const float* __restrict__ att_dst,
                         float* __restrict__ a_src, float* __restrict__ a_dst)
{
    int idx = blockIdx.x * blockDim.x + threadIdx.x;   // over B*N*H
    if (idx >= B_ * N_ * H_) return;
    int h  = idx % H_;
    int bn = idx / H_;
    const float* row = Hm + bn * HD_ + h * D_;
    const float* as  = att_src + h * D_;
    const float* ad  = att_dst + h * D_;
    float ss = 0.f, sd = 0.f;
#pragma unroll 8
    for (int d = 0; d < D_; ++d) { float v = row[d]; ss += v * as[d]; sd += v * ad[d]; }
    a_src[idx] = ss;
    a_dst[idx] = sd;
}

// ---------------- kernel 4: init m / denom / out ----------------
__global__ void k_init(unsigned* __restrict__ m, float* __restrict__ denom,
                       float* __restrict__ out, const float* __restrict__ bias)
{
    int i = blockIdx.x * blockDim.x + threadIdx.x;
    if (i < B_ * N_ * H_) { m[i] = ORD_NEG_INF; denom[i] = 0.0f; }
    if (i < B_ * N_ * HD_) out[i] = bias[i % HD_];
}

// ---------------- kernel 5: per-dst segment max ----------------
__global__ void k_edge_max(const int* __restrict__ src, const int* __restrict__ dst,
                           const float* __restrict__ a_src, const float* __restrict__ a_dst,
                           unsigned* __restrict__ m)
{
    int idx = blockIdx.x * blockDim.x + threadIdx.x;   // over ETOT_*B_*H_
    if (idx >= ETOT_ * B_ * H_) return;
    int e  = idx / (B_ * H_);
    int bh = idx % (B_ * H_);
    int b  = bh / H_, h = bh % H_;
    int s, d;
    if (e < E_) { s = src[e]; d = dst[e]; } else { s = d = e - E_; }
    float v = lrelu(a_src[(b * N_ + s) * H_ + h] + a_dst[(b * N_ + d) * H_ + h]);
    atomicMax(&m[(b * N_ + d) * H_ + h], f2ord(v));
}

// ---------------- kernel 6: per-dst exp-sum ----------------
__global__ void k_edge_sum(const int* __restrict__ src, const int* __restrict__ dst,
                           const float* __restrict__ a_src, const float* __restrict__ a_dst,
                           const unsigned* __restrict__ m, float* __restrict__ denom)
{
    int idx = blockIdx.x * blockDim.x + threadIdx.x;   // over ETOT_*B_*H_
    if (idx >= ETOT_ * B_ * H_) return;
    int e  = idx / (B_ * H_);
    int bh = idx % (B_ * H_);
    int b  = bh / H_, h = bh % H_;
    int s, d;
    if (e < E_) { s = src[e]; d = dst[e]; } else { s = d = e - E_; }
    int mi = (b * N_ + d) * H_ + h;
    float v  = lrelu(a_src[(b * N_ + s) * H_ + h] + a_dst[(b * N_ + d) * H_ + h]);
    float ex = __expf(v - ord2f(m[mi]));
    atomicAdd(&denom[mi], ex);
}

// ---------------- kernel 7: weighted scatter-add ----------------
// grid (ETOT_, B_), block 768: one thread per (h,d) component of one edge.
__global__ __launch_bounds__(HD_) void k_edge_agg(
    const int* __restrict__ src, const int* __restrict__ dst,
    const float* __restrict__ a_src, const float* __restrict__ a_dst,
    const unsigned* __restrict__ m, const float* __restrict__ denom,
    const float* __restrict__ Hm, float* __restrict__ out)
{
    int e = blockIdx.x;
    int b = blockIdx.y;
    int s, d;
    if (e < E_) { s = src[e]; d = dst[e]; } else { s = d = e - E_; }
    int tid = threadIdx.x;          // h*64 + dd
    int h   = tid / D_;
    int mi  = (b * N_ + d) * H_ + h;
    float v     = lrelu(a_src[(b * N_ + s) * H_ + h] + a_dst[(b * N_ + d) * H_ + h]);
    float alpha = __expf(v - ord2f(m[mi])) / (denom[mi] + 1e-16f);
    float hv    = Hm[(b * N_ + s) * HD_ + tid];
    atomicAdd(&out[(b * N_ + d) * HD_ + tid], alpha * hv);
}

extern "C" void kernel_launch(void* const* d_in, const int* in_sizes, int n_in,
                              void* d_out, int out_size, void* d_ws, size_t ws_size,
                              hipStream_t stream) {
    const float* x        = (const float*)d_in[0];   // [B,N,C]
    const float* W        = (const float*)d_in[1];   // [C,HD]
    const float* att_src  = (const float*)d_in[2];   // [H,D]
    const float* att_dst  = (const float*)d_in[3];   // [H,D]
    const float* bias     = (const float*)d_in[4];   // [HD]
    const int*   eidx     = (const int*)d_in[5];     // [2,E]
    const int*   src      = eidx;
    const int*   dst      = eidx + E_;
    float* out = (float*)d_out;                      // [B,N,HD]

    // workspace carve-up (256B aligned)
    char* base = (char*)d_ws;
    size_t off = 0;
    auto carve = [&](size_t bytes) -> char* {
        char* p = base + off;
        off = (off + bytes + 255) & ~(size_t)255;
        return p;
    };
    bf16_t*   x_bf   = (bf16_t*)  carve((size_t)M_ * C_ * sizeof(bf16_t));
    bf16_t*   W_pk   = (bf16_t*)  carve((size_t)C_ * HD_ * sizeof(bf16_t));
    float*    Hm     = (float*)   carve((size_t)M_ * HD_ * sizeof(float));
    float*    a_src  = (float*)   carve((size_t)B_ * N_ * H_ * sizeof(float));
    float*    a_dst  = (float*)   carve((size_t)B_ * N_ * H_ * sizeof(float));
    unsigned* m      = (unsigned*)carve((size_t)B_ * N_ * H_ * sizeof(unsigned));
    float*    denom  = (float*)   carve((size_t)B_ * N_ * H_ * sizeof(float));
    (void)ws_size;

    // 1) convert x to bf16; convert+pack W into fragment layout
    {
        int nx = M_ * C_;
        k_cvt_bf16<<<(nx + 255) / 256, 256, 0, stream>>>(x, x_bf, nx);
        int nw = C_ * HD_;
        k_pack_w<<<(nw + 255) / 256, 256, 0, stream>>>(W, W_pk);
    }

    // 2) h = x @ W via WMMA: 313 32-row blocks x 12 groups (32x64 per wave)
    {
        dim3 grid((M_ + 31) / 32, 3);   // (313, 3)
        dim3 block(32, 4);
        k_gemm_wmma<<<grid, block, 0, stream>>>(x_bf, W_pk, Hm);
    }

    // 3) attention scores
    {
        int n = B_ * N_ * H_;
        k_scores<<<(n + 255) / 256, 256, 0, stream>>>(Hm, att_src, att_dst, a_src, a_dst);
    }

    // 4) init m/denom/out(=bias)
    {
        int n = B_ * N_ * HD_;              // covers both ranges
        k_init<<<(n + 255) / 256, 256, 0, stream>>>(m, denom, out, bias);
    }

    // 5) segment max
    {
        int n = ETOT_ * B_ * H_;
        k_edge_max<<<(n + 255) / 256, 256, 0, stream>>>(src, dst, a_src, a_dst, m);
    }

    // 6) segment exp-sum
    {
        int n = ETOT_ * B_ * H_;
        k_edge_sum<<<(n + 255) / 256, 256, 0, stream>>>(src, dst, a_src, a_dst, m, denom);
    }

    // 7) weighted scatter-add
    {
        dim3 grid(ETOT_, B_);
        k_edge_agg<<<grid, HD_, 0, stream>>>(src, dst, a_src, a_dst, m, denom, Hm, out);
    }
}
